// AttnProcessor_11596411699930
// MI455X (gfx1250) — matrix operand order repaired and checked by reading, not compiled, not run on hardware
//
#include <hip/hip_runtime.h>
#include <hip/hip_bf16.h>

// ---------------- problem constants ----------------
#define B_  8
#define S_  4096
#define L_  77
#define LP_ 96          // L padded to 6 tiles of 16
#define D_  1280
#define CE_ 2048
#define H_  20
#define DH_ 64
#define SCALE_ 0.125f   // 64^-0.5

typedef __attribute__((ext_vector_type(16))) __bf16 v16bf;
typedef __attribute__((ext_vector_type(8)))  float  v8f;

union Frag16 { v16bf v; uint4 u[2]; };

// float -> bf16 round-to-nearest-even (as u16 payload)
static __device__ __forceinline__ unsigned short f2bf(float f) {
    unsigned int u = __float_as_uint(f);
    unsigned int r = u + 0x7FFFu + ((u >> 16) & 1u);
    return (unsigned short)(r >> 16);
}

// ordered-uint encoding for fp32 atomic max
static __device__ __forceinline__ unsigned int enc_f(float f) {
    unsigned int u = __float_as_uint(f);
    return (u & 0x80000000u) ? ~u : (u | 0x80000000u);
}
static __device__ __forceinline__ float dec_f(unsigned int u) {
    unsigned int b = (u & 0x80000000u) ? (u & 0x7FFFFFFFu) : ~u;
    return __uint_as_float(b);
}

static __device__ __forceinline__ v8f wmma_bf16(v16bf a, v16bf b, v8f c) {
    return __builtin_amdgcn_wmma_f32_16x16x32_bf16(false, a, false, b, (short)0, c, false, false);
}

// A-matrix 16x32 bf16 fragment. `row_kc` points at A[row*ld + kc] for this
// lane's row (row = lane&15). Per ISA layout: lanes<16 hold K 0..7 / 16..23,
// lanes>=16 hold K 8..15 / 24..31 -> two contiguous 16B loads.
static __device__ __forceinline__ v16bf load_a16(const unsigned short* row_kc, int lane) {
    Frag16 f;
    const int k0 = (lane >> 4) << 3;
    f.u[0] = *(const uint4*)(row_kc + k0);
    f.u[1] = *(const uint4*)(row_kc + k0 + 16);
    return f.v;
}

// B-matrix 32x16 bf16 fragment from a column-major column. `col_kc` points at
// Bt[n*K + kc] for this lane's column (n = lane&15). lanes<16 hold K 0..7 / 16..23,
// lanes>=16 hold K 8..15 / 24..31 -> two contiguous 16B loads.
static __device__ __forceinline__ v16bf load_b16(const unsigned short* col_kc, int lane) {
    Frag16 f;
    const int k0 = (lane >> 4) << 3;
    f.u[0] = *(const uint4*)(col_kc + k0);
    f.u[1] = *(const uint4*)(col_kc + k0 + 16);
    return f.v;
}

static __device__ __forceinline__ v16bf zero_frag16() {
    Frag16 f;
    f.u[0] = make_uint4(0u, 0u, 0u, 0u);
    f.u[1] = make_uint4(0u, 0u, 0u, 0u);
    return f.v;
}

// ---------------- casts ----------------
__global__ void cast_bf16_kernel(const float* __restrict__ src,
                                 unsigned short* __restrict__ dst, long n) {
    long i = (long)blockIdx.x * blockDim.x + threadIdx.x;
    if (i < n) dst[i] = f2bf(src[i]);
}

// W[K,N] row-major fp32 -> Wt[N,K] bf16 (column-major for B fragments)
__global__ void castT_bf16_kernel(const float* __restrict__ W,
                                  unsigned short* __restrict__ Wt, int K, int N) {
    long i = (long)blockIdx.x * blockDim.x + threadIdx.x;
    if (i < (long)K * N) {
        int k = (int)(i / N), n = (int)(i % N);
        Wt[(size_t)n * K + k] = f2bf(W[i]);
    }
}

// ---------------- dense GEMM: C = A[M,K] * Bt[N,K]^T ----------------
// Register-blocked: one wave owns a 64x64 output tile (4x4 subtiles of
// 16x16), so each K-step loads 4 A-frags + 4 B-frags and issues 16 WMMAs.
// Cbf!=null -> bf16 out; else fp32 out (+bias).
__global__ __launch_bounds__(32)
void gemm_bf16_kernel(const unsigned short* __restrict__ A,
                      const unsigned short* __restrict__ Bt,
                      int N, int K,
                      unsigned short* __restrict__ Cbf,
                      float* __restrict__ Cf,
                      const float* __restrict__ bias) {
    const int lane = threadIdx.x;
    const int n0 = blockIdx.x * 64;
    const int m0 = blockIdx.y * 64;

    const unsigned short* arow[4];
    const unsigned short* bcol[4];
#pragma unroll
    for (int i = 0; i < 4; ++i) {
        arow[i] = A  + (size_t)(m0 + 16 * i + (lane & 15)) * K;
        bcol[i] = Bt + (size_t)(n0 + 16 * i + (lane & 15)) * K;
    }

    v8f acc[4][4];
#pragma unroll
    for (int i = 0; i < 4; ++i)
#pragma unroll
        for (int j = 0; j < 4; ++j) acc[i][j] = (v8f){};

    for (int kc = 0; kc < K; kc += 32) {
        if (kc + 256 < K) __builtin_prefetch(arow[0] + kc + 256, 0, 0);
        v16bf a[4], b[4];
#pragma unroll
        for (int i = 0; i < 4; ++i) a[i] = load_a16(arow[i] + kc, lane);
#pragma unroll
        for (int j = 0; j < 4; ++j) b[j] = load_b16(bcol[j] + kc, lane);
#pragma unroll
        for (int i = 0; i < 4; ++i)
#pragma unroll
            for (int j = 0; j < 4; ++j)
                acc[i][j] = wmma_bf16(a[i], b[j], acc[i][j]);
    }

    const int rb = (lane >> 4) << 3;     // accumulator row base within subtile
    const int nc = lane & 15;            // column within subtile
#pragma unroll
    for (int i = 0; i < 4; ++i) {
#pragma unroll
        for (int j = 0; j < 4; ++j) {
            const int col   = n0 + 16 * j + nc;
            const int rbase = m0 + 16 * i + rb;
            if (Cbf) {
#pragma unroll
                for (int r = 0; r < 8; ++r)
                    Cbf[(size_t)(rbase + r) * N + col] = f2bf(acc[i][j][r]);
            } else {
                const float bv = bias[col];
#pragma unroll
                for (int r = 0; r < 8; ++r)
                    Cf[(size_t)(rbase + r) * N + col] = acc[i][j][r] + bv;
            }
        }
    }
}

// ---------------- K/V projection (batched, L padded to 96) ----------------
// 1x4 N-blocked: one A-frag reused across 4 column tiles.
// transposeStore==0: K_out[b][row][n] ; ==1: Vt_out[b][n][row]
__global__ __launch_bounds__(32)
void gemm_kv_kernel(const unsigned short* __restrict__ Enc,   // [B][77][2048]
                    const unsigned short* __restrict__ Wt,    // [1280][2048]
                    unsigned short* __restrict__ Dst,
                    int transposeStore) {
    const int lane = threadIdx.x;
    const int n0 = blockIdx.x * 64;    // 0..1216
    const int t  = blockIdx.y;         // 0..5 (rows of 16, padded region zeroed)
    const int b  = blockIdx.z;
    const int arow = t * 16 + (lane & 15);
    const bool avalid = arow < L_;
    const unsigned short* arp = Enc + ((size_t)b * L_ + arow) * CE_;
    const unsigned short* bcol[4];
#pragma unroll
    for (int j = 0; j < 4; ++j)
        bcol[j] = Wt + (size_t)(n0 + 16 * j + (lane & 15)) * CE_;

    v8f acc[4];
#pragma unroll
    for (int j = 0; j < 4; ++j) acc[j] = (v8f){};

    for (int kc = 0; kc < CE_; kc += 32) {
        v16bf a = avalid ? load_a16(arp + kc, lane) : zero_frag16();
#pragma unroll
        for (int j = 0; j < 4; ++j)
            acc[j] = wmma_bf16(a, load_b16(bcol[j] + kc, lane), acc[j]);
    }

    const int rb = (lane >> 4) << 3;
    const int nc = lane & 15;
    const int rbase = t * 16 + rb;
#pragma unroll
    for (int j = 0; j < 4; ++j) {
        const int col = n0 + 16 * j + nc;
        if (!transposeStore) {
            unsigned short* d = Dst + (size_t)b * LP_ * D_;
#pragma unroll
            for (int r = 0; r < 8; ++r)
                d[(size_t)(rbase + r) * D_ + col] = f2bf(acc[j][r]);
        } else {
            unsigned short* d = Dst + (size_t)b * D_ * LP_;
#pragma unroll
            for (int r = 0; r < 8; ++r)
                d[(size_t)col * LP_ + (rbase + r)] = f2bf(acc[j][r]);
        }
    }
}

// ---------------- global score max (masked to l<77) ----------------
__global__ __launch_bounds__(32)
void score_max_kernel(const unsigned short* __restrict__ Q,   // [B][S][1280]
                      const unsigned short* __restrict__ Kb,  // [B][96][1280]
                      unsigned int* __restrict__ gmax) {
    const int lane = threadIdx.x;
    const int st = blockIdx.x, h = blockIdx.y, b = blockIdx.z;
    const int s0 = st * 16;
    const unsigned short* qrow =
        Q + ((size_t)b * S_ + s0 + (lane & 15)) * D_ + h * DH_;
    v16bf aq0 = load_a16(qrow, lane);
    v16bf aq1 = load_a16(qrow + 32, lane);
    float mmax = -3.0e38f;
    for (int t = 0; t < 5; ++t) {   // tiles covering l=0..79; mask >=77
        const unsigned short* kcp =
            Kb + ((size_t)b * LP_ + t * 16 + (lane & 15)) * D_ + h * DH_;
        v8f acc = {};
        acc = wmma_bf16(aq0, load_b16(kcp, lane), acc);
        acc = wmma_bf16(aq1, load_b16(kcp + 32, lane), acc);
        const int lcol = t * 16 + (lane & 15);
        if (lcol < L_) {
#pragma unroll
            for (int r = 0; r < 8; ++r) mmax = fmaxf(mmax, acc[r] * SCALE_);
        }
    }
#pragma unroll
    for (int off = 16; off > 0; off >>= 1)
        mmax = fmaxf(mmax, __shfl_xor(mmax, off, 32));
    if (lane == 0) atomicMax(gmax, enc_f(mmax));
}

// ---------------- fused scores + bias + softmax + PV ----------------
__global__ __launch_bounds__(32)
void attn_kernel(const unsigned short* __restrict__ Q,    // [B][S][1280]
                 const unsigned short* __restrict__ Kb,   // [B][96][1280]
                 const unsigned short* __restrict__ Vt,   // [B][1280][96]
                 const float* __restrict__ region,        // [B][S][77]
                 const float* __restrict__ sigma,
                 const unsigned int* __restrict__ gmaxp,
                 unsigned short* __restrict__ Attn) {     // [B][S][1280] bf16
    __shared__ __align__(16) float          SC[16][LP_];
    __shared__ __align__(16) unsigned short P[16][LP_];
    __shared__ float rinv[16];

    const int lane = threadIdx.x;
    const int st = blockIdx.x, h = blockIdx.y, b = blockIdx.z;
    const int s0 = st * 16;
    const float bias_scale = log1pf(0.1f * sigma[0]) * dec_f(*gmaxp);

    // ---- scores via WMMA -> LDS ----
    const unsigned short* qrow =
        Q + ((size_t)b * S_ + s0 + (lane & 15)) * D_ + h * DH_;
    v16bf aq0 = load_a16(qrow, lane);
    v16bf aq1 = load_a16(qrow + 32, lane);
    const int ncol = lane & 15;
    const int rb   = (lane >> 4) << 3;
    for (int t = 0; t < 6; ++t) {
        const unsigned short* kcp =
            Kb + ((size_t)b * LP_ + t * 16 + ncol) * D_ + h * DH_;
        v8f acc = {};
        acc = wmma_bf16(aq0, load_b16(kcp, lane), acc);
        acc = wmma_bf16(aq1, load_b16(kcp + 32, lane), acc);
#pragma unroll
        for (int r = 0; r < 8; ++r) SC[rb + r][t * 16 + ncol] = acc[r] * SCALE_;
    }
    __syncthreads();

    // ---- per-row biased softmax (un-normalized probs, bf16) ----
    if (lane < 16) {
        const int m = lane;
        const float* reg = region + ((size_t)b * S_ + s0 + m) * L_;
        float rmax = -3.0e38f;
        for (int n = 0; n < L_; ++n) {
            float v = SC[m][n] + reg[n] * bias_scale;
            SC[m][n] = v;
            rmax = fmaxf(rmax, v);
        }
        float sum = 0.f;
        for (int n = 0; n < L_; ++n) {
            float e = __expf(SC[m][n] - rmax);
            sum += e;
            P[m][n] = f2bf(e);
        }
        for (int n = L_; n < LP_; ++n) P[m][n] = 0;
        rinv[m] = 1.0f / sum;
    }
    __syncthreads();

    // ---- PV: out[16x64] = P[16x96] * V[96x64], row-scaled by 1/sum ----
    for (int nt = 0; nt < 4; ++nt) {
        v8f acc = {};
        for (int kc = 0; kc < LP_; kc += 32) {
            Frag16 fa;
            const int k0 = kc + ((lane >> 4) << 3);
            fa.u[0] = *(const uint4*)&P[ncol][k0];
            fa.u[1] = *(const uint4*)&P[ncol][k0 + 16];
            const unsigned short* vcol =
                Vt + ((size_t)b * D_ + h * DH_ + nt * 16 + ncol) * LP_ + kc;
            acc = wmma_bf16(fa.v, load_b16(vcol, lane), acc);
        }
        const int col = h * DH_ + nt * 16 + ncol;
#pragma unroll
        for (int r = 0; r < 8; ++r) {
            float o = acc[r] * rinv[rb + r];
            Attn[((size_t)b * S_ + s0 + rb + r) * D_ + col] = f2bf(o);
        }
    }
}

// ---------------- host ----------------
extern "C" void kernel_launch(void* const* d_in, const int* in_sizes, int n_in,
                              void* d_out, int out_size, void* d_ws, size_t ws_size,
                              hipStream_t stream) {
    const float* hs     = (const float*)d_in[0];
    const float* enc    = (const float*)d_in[1];
    const float* region = (const float*)d_in[2];
    const float* Wq     = (const float*)d_in[3];
    const float* Wk     = (const float*)d_in[4];
    const float* Wv     = (const float*)d_in[5];
    const float* Wo     = (const float*)d_in[6];
    const float* bo     = (const float*)d_in[7];
    const float* sigma  = (const float*)d_in[8];
    float* out = (float*)d_out;

    // workspace carve-up (256B aligned)
    size_t off = 0;
    auto carve = [&](size_t bytes) -> char* {
        char* p = (char*)d_ws + off;
        off += (bytes + 255) & ~(size_t)255;
        return p;
    };
    const size_t nBSD = (size_t)B_ * S_ * D_;          // 41,943,040
    const size_t nEnc = (size_t)B_ * L_ * CE_;         // 1,261,568
    unsigned short* hs_bf   = (unsigned short*)carve(nBSD * 2);
    unsigned short* q_bf    = (unsigned short*)carve(nBSD * 2);
    unsigned short* attn_bf = (unsigned short*)carve(nBSD * 2);
    unsigned short* enc_bf  = (unsigned short*)carve(nEnc * 2);
    unsigned short* Wq_t    = (unsigned short*)carve((size_t)D_ * D_ * 2);
    unsigned short* Wk_t    = (unsigned short*)carve((size_t)CE_ * D_ * 2);
    unsigned short* Wv_t    = (unsigned short*)carve((size_t)CE_ * D_ * 2);
    unsigned short* Wo_t    = (unsigned short*)carve((size_t)D_ * D_ * 2);
    unsigned short* k_bf    = (unsigned short*)carve((size_t)B_ * LP_ * D_ * 2);
    unsigned short* vt_bf   = (unsigned short*)carve((size_t)B_ * D_ * LP_ * 2);
    unsigned int*   gmax    = (unsigned int*)carve(256);

    // 1) casts
    cast_bf16_kernel<<<(unsigned)((nBSD + 255) / 256), 256, 0, stream>>>(hs, hs_bf, (long)nBSD);
    cast_bf16_kernel<<<(unsigned)((nEnc + 255) / 256), 256, 0, stream>>>(enc, enc_bf, (long)nEnc);
    {
        long n = (long)D_ * D_;
        castT_bf16_kernel<<<(unsigned)((n + 255) / 256), 256, 0, stream>>>(Wq, Wq_t, D_, D_);
        castT_bf16_kernel<<<(unsigned)((n + 255) / 256), 256, 0, stream>>>(Wo, Wo_t, D_, D_);
        long n2 = (long)CE_ * D_;
        castT_bf16_kernel<<<(unsigned)((n2 + 255) / 256), 256, 0, stream>>>(Wk, Wk_t, CE_, D_);
        castT_bf16_kernel<<<(unsigned)((n2 + 255) / 256), 256, 0, stream>>>(Wv, Wv_t, CE_, D_);
    }

    // 2) Q projection: [32768,1280] x [1280,1280], 64x64 tiles per wave
    gemm_bf16_kernel<<<dim3(D_ / 64, (B_ * S_) / 64), 32, 0, stream>>>(
        hs_bf, Wq_t, D_, D_, q_bf, nullptr, nullptr);

    // 3) K/V projections (batched, padded L)
    gemm_kv_kernel<<<dim3(D_ / 64, LP_ / 16, B_), 32, 0, stream>>>(enc_bf, Wk_t, k_bf, 0);
    gemm_kv_kernel<<<dim3(D_ / 64, LP_ / 16, B_), 32, 0, stream>>>(enc_bf, Wv_t, vt_bf, 1);

    // 4) global score max
    hipMemsetAsync(gmax, 0, sizeof(unsigned int), stream);
    score_max_kernel<<<dim3(S_ / 16, H_, B_), 32, 0, stream>>>(q_bf, k_bf, gmax);

    // 5) fused scores + bias + softmax + PV
    attn_kernel<<<dim3(S_ / 16, H_, B_), 32, 0, stream>>>(
        q_bf, k_bf, vt_bf, region, sigma, gmax, attn_bf);

    // 6) output projection + bias -> fp32 d_out, 64x64 tiles per wave
    gemm_bf16_kernel<<<dim3(D_ / 64, (B_ * S_) / 64), 32, 0, stream>>>(
        attn_bf, Wo_t, D_, D_, nullptr, out, bo);
}
